// GraphNetwork_49469433315461
// MI455X (gfx1250) — matrix operand order repaired and checked by reading, not compiled, not run on hardware
//
#include <hip/hip_runtime.h>
#include <stdint.h>

// ---------------------------------------------------------------------------
// Problem constants (fixed by the reference)
// ---------------------------------------------------------------------------
#define F_IN   256
#define HS     256
#define DH     512
#define MEMD   1024
#define BSZ    32
#define TLEN   512
#define NTOK   16384      /* BSZ*TLEN */
#define NEDGE  262144
#define NREL   8
#define G4     2048       /* 4*DH */

typedef __attribute__((ext_vector_type(16))) __bf16 v16bf;
typedef __attribute__((ext_vector_type(8)))  float  v8f;

__device__ __forceinline__ unsigned short f2bf(float f) {
  unsigned u = __float_as_uint(f);
  u += 0x7FFFu + ((u >> 16) & 1u);      // round-to-nearest-even
  return (unsigned short)(u >> 16);
}

union FragBF { v16bf v; uint4 q[2]; };

// A fragment (16x32 bf16).  ISA layout: lane-half h holds K = {h*8..h*8+7} U
// {16+h*8..16+h*8+7}.  Caller passes p = rowBase + k0 + h*8.
__device__ __forceinline__ v16bf load_fragA(const unsigned short* p) {
  FragBF f;
  f.q[0] = *(const uint4*)(p);
  f.q[1] = *(const uint4*)(p + 16);
  return f.v;
}
// B fragment (32x16 bf16, weights stored transposed: N-major rows of K).
// Lane-half h holds K = {h*8..h*8+7} U {16+h*8..16+h*8+7}.  p = rowBase + k0 + h*8.
__device__ __forceinline__ v16bf load_fragB(const unsigned short* p) {
  FragBF f;
  f.q[0] = *(const uint4*)(p);
  f.q[1] = *(const uint4*)(p + 16);
  return f.v;
}

__device__ __forceinline__ float sigf(float x) { return 1.0f / (1.0f + __expf(-x)); }

#define WMMA_BF16(a, b, c) \
  __builtin_amdgcn_wmma_f32_16x16x32_bf16(false, (a), false, (b), (short)0, (c), false, false)

// Async DMA of one 16B chunk per lane from global into LDS (ASYNCcnt-tracked).
__device__ __forceinline__ void async_ldsb128(unsigned ldsOff, const void* gaddr) {
  asm volatile("global_load_async_to_lds_b128 %0, %1, off"
               :: "v"(ldsOff), "v"((unsigned long long)(uintptr_t)gaddr)
               : "memory");
}
__device__ __forceinline__ void wait_async0() {
  asm volatile("s_wait_asynccnt 0" ::: "memory");
}

// ---------------------------------------------------------------------------
// Generic WMMA GEMM:  C[M,N] = act( (A[M,K] x Bt[N,K]^T) (+C) (+bias) (*cmask) )
// A,Bt bf16 (as u16), C f32.
// Block = 256 thr = 8 waves arranged 2(M) x 4(N): block tile 64 x 256.
// The 64x32 A block-tile is staged into LDS with GLOBAL_LOAD_ASYNC_TO_LDS_B128
// (one async-b128 per wave per k-step, double-buffered, s_wait_asynccnt +
// barrier), cutting A global traffic 4x.  B fragments double-buffered in
// registers from global.  The k-loop is unrolled 2x with ping/pong register
// sets so loads land directly in their consumers (no copies -> partial
// counter waits instead of full drains).  8 back-to-back WMMAs per k-step.
// grid.z = batch with element strides sA/sB/sC/sBias/sMask.
// ---------------------------------------------------------------------------
#define APAD 40   /* LDS row stride (halves): 80B -> conflict-free frag reads */
__global__ __launch_bounds__(256) void gemm_bf16_wmma(
    const unsigned short* __restrict__ A, int lda, long long sA,
    const unsigned short* __restrict__ Bt, int ldb, long long sB,
    float* __restrict__ C, int ldc, long long sC,
    const float* __restrict__ bias, long long sBias,
    const float* __restrict__ cmask, long long sMask,
    int M, int K, int act, int accum)
{
  __shared__ unsigned short aTile[2][64 * APAD];

  const int tid  = threadIdx.x;
  const int lane = tid & 31;
  const int wv   = tid >> 5;                   // 0..7
  const int m16  = lane & 15;
  const int hh   = lane >> 4;
  const int mL   = (wv & 1) * 32;              // wave tile row within block tile
  const int mT   = blockIdx.y * 64 + mL;       // global row
  const int nT   = blockIdx.x * 256 + (wv >> 1) * 64;
  const int zb   = blockIdx.z;
  A  += (long long)zb * sA;
  Bt += (long long)zb * sB;
  C  += (long long)zb * sC;
  const float* bp = bias  ? bias  + (long long)zb * sBias : (const float*)0;
  const float* mp = cmask ? cmask + (long long)zb * sMask : (const float*)0;

  // staging map: thread -> one 16B chunk of the 64x32 A tile
  const int srow = tid >> 2;                   // 0..63
  const int sseg = (tid & 3) * 8;              // half-offset within the 32-wide row
  const unsigned short* gA = A + (size_t)(blockIdx.y * 64 + srow) * lda + sseg;
  const unsigned ldsA0 = (unsigned)(uintptr_t)(&aTile[0][srow * APAD + sseg]);
  const unsigned ldsA1 = (unsigned)(uintptr_t)(&aTile[1][srow * APAD + sseg]);

  const unsigned short* bR0 = Bt + (size_t)(nT +  0 + m16) * ldb + hh * 8;
  const unsigned short* bR1 = Bt + (size_t)(nT + 16 + m16) * ldb + hh * 8;
  const unsigned short* bR2 = Bt + (size_t)(nT + 32 + m16) * ldb + hh * 8;
  const unsigned short* bR3 = Bt + (size_t)(nT + 48 + m16) * ldb + hh * 8;

  const unsigned short* aLb0 = &aTile[0][(mL + m16) * APAD + hh * 8];
  const unsigned short* aLb1 = &aTile[1][(mL + m16) * APAD + hh * 8];

  v8f acc[8];
#pragma unroll
  for (int i = 0; i < 8; ++i) acc[i] = (v8f){0.f,0.f,0.f,0.f,0.f,0.f,0.f,0.f};

#define WMMA8(A0_, A1_, B0_, B1_, B2_, B3_)          \
  do {                                               \
    acc[0] = WMMA_BF16(A0_, B0_, acc[0]);            \
    acc[1] = WMMA_BF16(A0_, B1_, acc[1]);            \
    acc[2] = WMMA_BF16(A0_, B2_, acc[2]);            \
    acc[3] = WMMA_BF16(A0_, B3_, acc[3]);            \
    acc[4] = WMMA_BF16(A1_, B0_, acc[4]);            \
    acc[5] = WMMA_BF16(A1_, B1_, acc[5]);            \
    acc[6] = WMMA_BF16(A1_, B2_, acc[6]);            \
    acc[7] = WMMA_BF16(A1_, B3_, acc[7]);            \
  } while (0)

  // prologue: stage k-step 0 into buf0, load B set P for k-step 0
  async_ldsb128(ldsA0, gA);
  v16bf b0 = load_fragB(bR0);
  v16bf b1 = load_fragB(bR1);
  v16bf b2 = load_fragB(bR2);
  v16bf b3 = load_fragB(bR3);
  wait_async0();
  __syncthreads();
  v16bf a0 = load_fragA(aLb0);
  v16bf a1 = load_fragA(aLb0 + 16 * APAD);

  // main loop: two 32-wide k-steps per iteration (K/32 is even for all uses)
  for (int k0 = 32; k0 + 32 < K; k0 += 64) {
    // even step: stage s+1 -> buf1, load B set Q, compute on set P
    async_ldsb128(ldsA1, gA + k0);
    v16bf c0 = load_fragB(bR0 + k0);
    v16bf c1 = load_fragB(bR1 + k0);
    v16bf c2 = load_fragB(bR2 + k0);
    v16bf c3 = load_fragB(bR3 + k0);
    WMMA8(a0, a1, b0, b1, b2, b3);
    wait_async0();
    __syncthreads();
    v16bf d0 = load_fragA(aLb1);
    v16bf d1 = load_fragA(aLb1 + 16 * APAD);
    // odd step: stage s+2 -> buf0, load B set P, compute on set Q
    async_ldsb128(ldsA0, gA + k0 + 32);
    b0 = load_fragB(bR0 + k0 + 32);
    b1 = load_fragB(bR1 + k0 + 32);
    b2 = load_fragB(bR2 + k0 + 32);
    b3 = load_fragB(bR3 + k0 + 32);
    WMMA8(d0, d1, c0, c1, c2, c3);
    wait_async0();
    __syncthreads();
    a0 = load_fragA(aLb0);
    a1 = load_fragA(aLb0 + 16 * APAD);
  }

  // epilogue: last two k-steps (set P holds step S-2; step S-1 at K-32)
  {
    const int kl = K - 32;
    async_ldsb128(ldsA1, gA + kl);
    v16bf c0 = load_fragB(bR0 + kl);
    v16bf c1 = load_fragB(bR1 + kl);
    v16bf c2 = load_fragB(bR2 + kl);
    v16bf c3 = load_fragB(bR3 + kl);
    WMMA8(a0, a1, b0, b1, b2, b3);
    wait_async0();
    __syncthreads();
    v16bf d0 = load_fragA(aLb1);
    v16bf d1 = load_fragA(aLb1 + 16 * APAD);
    WMMA8(d0, d1, c0, c1, c2, c3);
  }
#undef WMMA8

#pragma unroll
  for (int i = 0; i < 2; ++i) {
#pragma unroll
    for (int j = 0; j < 4; ++j) {
      const int nn = nT + j * 16 + m16;
#pragma unroll
      for (int r = 0; r < 8; ++r) {
        const int mm = mT + i * 16 + r + hh * 8;   // C layout: lanes16-31 -> M=r+8
        const size_t idx = (size_t)mm * ldc + nn;
        float v = acc[i * 4 + j][r];
        if (accum) v += C[idx];
        if (bp) v += bp[nn];
        if (mp) v *= mp[nn];
        if (act == 1)      v = fmaxf(v, 0.0f);
        else if (act == 2) v = tanhf(v);
        C[idx] = v;
      }
    }
  }
}

// ---------------------------------------------------------------------------
// LSTM recurrence scan.  One 1024-thread block per direction (grid.x = 2).
// h (32x512 bf16) lives in LDS (row stride 520 to spread banks); c in regs.
// Per step: g = gpre[t] + h @ Whh^T via WMMA (each wave owns all 4 gate tiles
// of its 32 hidden columns -> nonlinearity needs no cross-wave traffic),
// next step's gpre slice is prefetched (global_prefetch_b8) during compute.
// ---------------------------------------------------------------------------
#define HPAD 520
__global__ __launch_bounds__(1024) void lstm_scan(
    const float* __restrict__ preF, const float* __restrict__ preR,
    const unsigned short* __restrict__ whhF, const unsigned short* __restrict__ whhR,
    float* __restrict__ out, int T)
{
  const int dir = blockIdx.x;
  const float* gpre = dir ? preR : preF;
  const unsigned short* whh = dir ? whhR : whhF;
  const int outOff = dir * DH;

  __shared__ unsigned short hbuf[BSZ * HPAD];

  const int tid  = threadIdx.x;
  const int lane = tid & 31;
  const int wv   = tid >> 5;             // 0..31
  const int mh   = wv & 1;               // batch half (rows mh*16..+15)
  const int slot = wv >> 1;              // 0..15 -> 32 hidden cols
  const int j0   = slot * 32;
  const int m16  = lane & 15;
  const int hh   = lane >> 4;

  float c[16];
#pragma unroll
  for (int i = 0; i < 16; ++i) c[i] = 0.f;
  for (int i = tid; i < BSZ * HPAD; i += 1024) hbuf[i] = 0;
  __syncthreads();

  const unsigned short* aRow = &hbuf[(mh * 16 + m16) * HPAD + hh * 8];
  const unsigned short* bR[8];
#pragma unroll
  for (int gi = 0; gi < 4; ++gi)
#pragma unroll
    for (int q = 0; q < 2; ++q)
      bR[gi * 2 + q] = whh + (size_t)(gi * DH + j0 + q * 16 + m16) * DH + hh * 8;

  const v8f vzero = (v8f){0.f,0.f,0.f,0.f,0.f,0.f,0.f,0.f};

  for (int step = 0; step < T; ++step) {
    const int t = dir ? (T - 1 - step) : step;

    // prefetch next step's gate pre-activations while this step computes
    const int tn = dir ? (t - 1) : (t + 1);
    if (tn >= 0 && tn < T) {
      const float* gpn = gpre + (size_t)tn * BSZ * G4 +
                         (size_t)(mh * 16 + hh * 8) * G4 + j0 + m16;
#pragma unroll
      for (int gi = 0; gi < 4; ++gi)
        __builtin_prefetch(gpn + gi * DH, 0, 1);
    }

    v8f acc[8];
#pragma unroll
    for (int i = 0; i < 8; ++i) acc[i] = vzero;

    for (int k0 = 0; k0 < DH; k0 += 32) {
      v16bf a = load_fragA(aRow + k0);
      v16bf bb[4];
#pragma unroll
      for (int i = 0; i < 4; ++i) bb[i] = load_fragB(bR[i] + k0);
#pragma unroll
      for (int i = 0; i < 4; ++i) {
        v16bf nb = load_fragB(bR[i + 4] + k0);   // pipeline second half
        acc[i] = WMMA_BF16(a, bb[i], acc[i]);
        bb[i] = nb;
      }
#pragma unroll
      for (int i = 0; i < 4; ++i)
        acc[i + 4] = WMMA_BF16(a, bb[i], acc[i + 4]);
    }
    __syncthreads();   // all reads of hbuf done before overwriting h

    const float* gp = gpre + (size_t)t * BSZ * G4;
    float* op = out + (size_t)t * BSZ * MEMD + outOff;
#pragma unroll
    for (int r = 0; r < 8; ++r) {
      const int bb2 = mh * 16 + r + hh * 8;
#pragma unroll
      for (int q = 0; q < 2; ++q) {
        const int j = j0 + q * 16 + m16;
        const float* g = gp + (size_t)bb2 * G4 + j;
        const float gi_ = acc[0 * 2 + q][r] + g[0];
        const float gf_ = acc[1 * 2 + q][r] + g[DH];
        const float gg_ = acc[2 * 2 + q][r] + g[2 * DH];
        const float go_ = acc[3 * 2 + q][r] + g[3 * DH];
        const float cc = c[r * 2 + q];
        const float cn = sigf(gf_) * cc + sigf(gi_) * tanhf(gg_);
        const float hn = sigf(go_) * tanhf(cn);
        c[r * 2 + q] = cn;
        hbuf[bb2 * HPAD + j] = f2bf(hn);
        op[(size_t)bb2 * MEMD + j] = hn;
      }
    }
    __syncthreads();
  }
}

// ---------------------------------------------------------------------------
// Small utility / streaming kernels
// ---------------------------------------------------------------------------
__global__ void fill_f32(float* __restrict__ p, float v, long long n) {
  long long i = (long long)blockIdx.x * blockDim.x + threadIdx.x;
  if (i < n) p[i] = v;
}
__global__ void cast_bf(const float* __restrict__ s, unsigned short* __restrict__ d, long long n) {
  long long i = (long long)blockIdx.x * blockDim.x + threadIdx.x;
  if (i < n) d[i] = f2bf(s[i]);
}
// src (z,R,C) row-major -> dst (z,C,R)
__global__ void cast_bf_tr(const float* __restrict__ s, unsigned short* __restrict__ d, int Rr, int Cc) {
  int z = blockIdx.z;
  int i = blockIdx.x * blockDim.x + threadIdx.x;
  if (i < Rr * Cc) {
    int r = i / Cc, cI = i % Cc;
    d[(size_t)z * Rr * Cc + (size_t)cI * Rr + r] = f2bf(s[(size_t)z * Rr * Cc + i]);
  }
}
__global__ void add_vec(const float* __restrict__ a, const float* __restrict__ b,
                        float* __restrict__ o, int n) {
  int i = blockIdx.x * blockDim.x + threadIdx.x;
  if (i < n) o[i] = a[i] + b[i];
}
__global__ void count_edges(const int* __restrict__ dst, const int* __restrict__ et,
                            float* __restrict__ cnt, int E) {
  int e = blockIdx.x * blockDim.x + threadIdx.x;
  if (e < E) atomicAdd(&cnt[(size_t)dst[e] * NREL + et[e]], 1.0f);
}
// h1[dst] += xW[src, etype] * 1/max(cnt,1)   (4 floats per thread)
__global__ void scatter_msg(const float* __restrict__ xW, const int* __restrict__ src,
                            const int* __restrict__ dst, const int* __restrict__ et,
                            const float* __restrict__ cnt, float* __restrict__ h1, int E) {
  long long i = (long long)blockIdx.x * blockDim.x + threadIdx.x;
  int e = (int)(i >> 6);
  if (e >= E) return;
  int c4 = ((int)i & 63) * 4;
  int s = src[e], d = dst[e], r = et[e];
  float sc = 1.0f / fmaxf(cnt[(size_t)d * NREL + r], 1.0f);
  const float* x = xW + (size_t)s * (NREL * HS) + (size_t)r * HS + c4;
  float* o = h1 + (size_t)d * HS + c4;
  atomicAdd(o + 0, x[0] * sc); atomicAdd(o + 1, x[1] * sc);
  atomicAdd(o + 2, x[2] * sc); atomicAdd(o + 3, x[3] * sc);
}
__global__ void scatter_agg(const float* __restrict__ h1, const int* __restrict__ src,
                            const int* __restrict__ dst, float* __restrict__ agg, int E) {
  long long i = (long long)blockIdx.x * blockDim.x + threadIdx.x;
  int e = (int)(i >> 6);
  if (e >= E) return;
  int c4 = ((int)i & 63) * 4;
  int s = src[e], d = dst[e];
  const float* x = h1 + (size_t)s * HS + c4;
  float* o = agg + (size_t)d * HS + c4;
  atomicAdd(o + 0, x[0]); atomicAdd(o + 1, x[1]);
  atomicAdd(o + 2, x[2]); atomicAdd(o + 3, x[3]);
}
// conv[t,b,:] = bf16(concat(features,h2)[row n=b*T+t])
__global__ void build_conv(const float* __restrict__ feat, const float* __restrict__ h2,
                           unsigned short* __restrict__ conv) {
  long long i = (long long)blockIdx.x * blockDim.x + threadIdx.x;
  if (i >= (long long)NTOK * DH) return;
  int k = (int)(i & (DH - 1));
  long long row = i >> 9;                 // t*B + b
  int t = (int)(row >> 5), b = (int)(row & 31);
  size_t n = (size_t)b * TLEN + t;
  float v = (k < F_IN) ? feat[n * F_IN + k] : h2[n * HS + (k - F_IN)];
  conv[i] = f2bf(v);
}
// Mm[t,b,m] = bf16(M[t,b,m] * umask[b,t])
__global__ void cast_mask(const float* __restrict__ M, const float* __restrict__ um,
                          unsigned short* __restrict__ d) {
  long long i = (long long)blockIdx.x * blockDim.x + threadIdx.x;
  if (i >= (long long)NTOK * MEMD) return;
  long long row = i >> 10;
  int t = (int)(row >> 5), b = (int)(row & 31);
  d[i] = f2bf(M[i] * um[b * TLEN + t]);
}
// Ms[b,m,s] = bf16(M[s,b,m])   (per-batch transpose for the context GEMM)
__global__ void build_Ms(const float* __restrict__ M, unsigned short* __restrict__ d) {
  long long i = (long long)blockIdx.x * blockDim.x + threadIdx.x;
  if (i >= (long long)BSZ * MEMD * TLEN) return;
  int b = (int)(i / ((long long)MEMD * TLEN));
  long long rem = i - (long long)b * MEMD * TLEN;
  int m = (int)(rem / TLEN);
  int s = (int)(rem % TLEN);
  d[i] = f2bf(M[((size_t)s * BSZ + b) * MEMD + m]);
}
// alpha = (exp(v-max)*mask) / sum(exp(v-max)*mask), one 512-thr block per row.
__global__ __launch_bounds__(512) void softmax_row(const float* __restrict__ sc,
                                                   const float* __restrict__ um,
                                                   unsigned short* __restrict__ alpha) {
  const int row = blockIdx.x;             // b*TLEN + t
  const int b = row >> 9;
  const int s = threadIdx.x;
  const float x = sc[(size_t)row * TLEN + s];
  __shared__ float red[16];
  float m = x;
  for (int o = 16; o > 0; o >>= 1) m = fmaxf(m, __shfl_xor(m, o, 32));
  if ((s & 31) == 0) red[s >> 5] = m;
  __syncthreads();
  float mm = red[0];
#pragma unroll
  for (int i = 1; i < 16; ++i) mm = fmaxf(mm, red[i]);
  float e = __expf(x - mm) * um[b * TLEN + s];
  float sum = e;
  for (int o = 16; o > 0; o >>= 1) sum += __shfl_xor(sum, o, 32);
  __syncthreads();
  if ((s & 31) == 0) red[s >> 5] = sum;
  __syncthreads();
  float tot = 0.f;
#pragma unroll
  for (int i = 0; i < 16; ++i) tot += red[i];
  alpha[(size_t)row * TLEN + s] = f2bf(e / fmaxf(tot, 1e-30f));
}

// ---------------------------------------------------------------------------
// Host orchestration
// ---------------------------------------------------------------------------
extern "C" void kernel_launch(void* const* d_in, const int* in_sizes, int n_in,
                              void* d_out, int out_size, void* d_ws, size_t ws_size,
                              hipStream_t stream)
{
  (void)in_sizes; (void)n_in; (void)out_size; (void)ws_size;
  const float* features = (const float*)d_in[0];
  const float* umask    = (const float*)d_in[1];
  const float* W_rgcn   = (const float*)d_in[2];
  const float* W_root   = (const float*)d_in[3];
  const float* b_rgcn   = (const float*)d_in[4];
  const float* Wg_rel   = (const float*)d_in[5];
  const float* bg_rel   = (const float*)d_in[6];
  const float* Wg_root  = (const float*)d_in[7];
  const float* w_ih_l0  = (const float*)d_in[8];
  const float* w_hh_l0  = (const float*)d_in[9];
  const float* b_ih_l0  = (const float*)d_in[10];
  const float* b_hh_l0  = (const float*)d_in[11];
  const float* w_ih_l0r = (const float*)d_in[12];
  const float* w_hh_l0r = (const float*)d_in[13];
  const float* b_ih_l0r = (const float*)d_in[14];
  const float* b_hh_l0r = (const float*)d_in[15];
  const float* w_ih_l1  = (const float*)d_in[16];
  const float* w_hh_l1  = (const float*)d_in[17];
  const float* b_ih_l1  = (const float*)d_in[18];
  const float* b_hh_l1  = (const float*)d_in[19];
  const float* w_ih_l1r = (const float*)d_in[20];
  const float* w_hh_l1r = (const float*)d_in[21];
  const float* b_ih_l1r = (const float*)d_in[22];
  const float* b_hh_l1r = (const float*)d_in[23];
  const float* W_att    = (const float*)d_in[24];
  const float* b_att    = (const float*)d_in[25];
  const float* W_lin    = (const float*)d_in[26];
  const float* b_lin    = (const float*)d_in[27];
  const int*   e_src    = (const int*)d_in[28];
  const int*   e_dst    = e_src + NEDGE;
  const int*   e_type   = (const int*)d_in[29];

  const size_t MB = (size_t)1 << 20;
  uint8_t* ws = (uint8_t*)d_ws;
  size_t off = 0;
  auto carve = [&](size_t bytes) -> uint8_t* {
    uint8_t* p = ws + off;
    off = (off + bytes + 255) & ~(size_t)255;
    return p;
  };

  // bf16 weight copies (persistent for the launch)
  unsigned short* wrgcnT = (unsigned short*)carve((size_t)G4 * F_IN * 2);   // (r*HS+h, f)
  unsigned short* wrootT = (unsigned short*)carve((size_t)HS * F_IN * 2);   // (h, f)
  unsigned short* wgrel  = (unsigned short*)carve((size_t)HS * HS * 2);
  unsigned short* wgroot = (unsigned short*)carve((size_t)HS * HS * 2);
  unsigned short* wih0   = (unsigned short*)carve((size_t)G4 * DH * 2);
  unsigned short* whh0   = (unsigned short*)carve((size_t)G4 * DH * 2);
  unsigned short* wih0r  = (unsigned short*)carve((size_t)G4 * DH * 2);
  unsigned short* whh0r  = (unsigned short*)carve((size_t)G4 * DH * 2);
  unsigned short* wih1   = (unsigned short*)carve((size_t)G4 * MEMD * 2);
  unsigned short* whh1   = (unsigned short*)carve((size_t)G4 * DH * 2);
  unsigned short* wih1r  = (unsigned short*)carve((size_t)G4 * MEMD * 2);
  unsigned short* whh1r  = (unsigned short*)carve((size_t)G4 * DH * 2);
  unsigned short* watt   = (unsigned short*)carve((size_t)MEMD * MEMD * 2);
  unsigned short* wlin   = (unsigned short*)carve((size_t)DH * MEMD * 2);
  float* bias0  = (float*)carve(G4 * 4);
  float* bias0r = (float*)carve(G4 * 4);
  float* bias1  = (float*)carve(G4 * 4);
  float* bias1r = (float*)carve(G4 * 4);

  // big reusable regions
  uint8_t* P0 = carve(128 * MB);   // xW -> pre_f(l0) -> pre_f(l1) -> Xt | scores | alpha
  uint8_t* P1 = carve(128 * MB);   // pre_r(l0) -> pre_r(l1) -> Xtbf | Mbf | Mmbf | Msbf
  uint8_t* DR = carve(96 * MB);    // rgcn scratch -> l0out | l0bf
  uint8_t* ER = carve(160 * MB);   // M | att | attbf

  float*          xW      = (float*)P0;
  float*          preF    = (float*)P0;
  float*          Xt      = (float*)P0;
  float*          scores  = (float*)(P0 + 64 * MB);
  unsigned short* alphabf = (unsigned short*)(P0 + 96 * MB);
  float*          preR    = (float*)P1;
  unsigned short* Xtbf    = (unsigned short*)P1;
  unsigned short* Mbf     = (unsigned short*)(P1 + 32 * MB);
  unsigned short* Mmbf    = (unsigned short*)(P1 + 64 * MB);
  unsigned short* Msbf    = (unsigned short*)(P1 + 96 * MB);
  unsigned short* featbf  = (unsigned short*)DR;
  float*          cnt     = (float*)(DR + 8 * MB);
  float*          h1      = (float*)(DR + 9 * MB);
  float*          agg     = (float*)(DR + 25 * MB);
  unsigned short* h1bf    = (unsigned short*)(DR + 41 * MB);
  unsigned short* aggbf   = (unsigned short*)(DR + 49 * MB);
  float*          h2      = (float*)(DR + 57 * MB);
  unsigned short* convbf  = (unsigned short*)(DR + 73 * MB);
  float*          l0out   = (float*)DR;                       // reuse after conv
  unsigned short* l0bf    = (unsigned short*)(DR + 64 * MB);
  float*          Mbuf    = (float*)ER;
  float*          attb    = (float*)(ER + 64 * MB);
  unsigned short* attbf   = (unsigned short*)(ER + 128 * MB);
  float*          outp    = (float*)d_out;

  auto castN = [&](const float* s, unsigned short* d, long long n) {
    cast_bf<<<dim3((unsigned)((n + 255) / 256)), dim3(256), 0, stream>>>(s, d, n);
  };
  auto gemm = [&](const unsigned short* A, int lda, long long sA,
                  const unsigned short* Bt, int ldb, long long sB,
                  float* C, int ldc, long long sC,
                  const float* bias, long long sb,
                  const float* cm, long long sm,
                  int M, int N, int K, int act, int accum, int nb) {
    dim3 g((unsigned)(N / 256), (unsigned)((M + 63) / 64), (unsigned)nb);
    gemm_bf16_wmma<<<g, dim3(256), 0, stream>>>(A, lda, sA, Bt, ldb, sB, C, ldc, sC,
                                                bias, sb, cm, sm, M, K, act, accum);
  };

  // ---- weight casts ----
  cast_bf_tr<<<dim3(256, 1, NREL), dim3(256), 0, stream>>>(W_rgcn, wrgcnT, F_IN, HS);
  cast_bf_tr<<<dim3(256, 1, 1), dim3(256), 0, stream>>>(W_root, wrootT, F_IN, HS);
  castN(Wg_rel,   wgrel,  (long long)HS * HS);
  castN(Wg_root,  wgroot, (long long)HS * HS);
  castN(w_ih_l0,  wih0,  (long long)G4 * DH);
  castN(w_hh_l0,  whh0,  (long long)G4 * DH);
  castN(w_ih_l0r, wih0r, (long long)G4 * DH);
  castN(w_hh_l0r, whh0r, (long long)G4 * DH);
  castN(w_ih_l1,  wih1,  (long long)G4 * MEMD);
  castN(w_hh_l1,  whh1,  (long long)G4 * DH);
  castN(w_ih_l1r, wih1r, (long long)G4 * MEMD);
  castN(w_hh_l1r, whh1r, (long long)G4 * DH);
  castN(W_att,    watt,  (long long)MEMD * MEMD);
  castN(W_lin,    wlin,  (long long)DH * MEMD);
  add_vec<<<dim3(8), dim3(256), 0, stream>>>(b_ih_l0,  b_hh_l0,  bias0,  G4);
  add_vec<<<dim3(8), dim3(256), 0, stream>>>(b_ih_l0r, b_hh_l0r, bias0r, G4);
  add_vec<<<dim3(8), dim3(256), 0, stream>>>(b_ih_l1,  b_hh_l1,  bias1,  G4);
  add_vec<<<dim3(8), dim3(256), 0, stream>>>(b_ih_l1r, b_hh_l1r, bias1r, G4);

  // ---- RGCN ----
  castN(features, featbf, (long long)NTOK * F_IN);
  gemm(featbf, F_IN, 0, wrgcnT, F_IN, 0, xW, G4, 0, 0, 0, 0, 0,
       NTOK, G4, F_IN, 0, 0, 1);                                       // xW (N, R*HS)
  fill_f32<<<dim3(512), dim3(256), 0, stream>>>(cnt, 0.f, (long long)NTOK * NREL);
  count_edges<<<dim3(NEDGE / 256), dim3(256), 0, stream>>>(e_dst, e_type, cnt, NEDGE);
  gemm(featbf, F_IN, 0, wrootT, F_IN, 0, h1, HS, 0, b_rgcn, 0, 0, 0,
       NTOK, HS, F_IN, 0, 0, 1);                                       // h1 = x@W_root + b
  scatter_msg<<<dim3((NEDGE * 64) / 256), dim3(256), 0, stream>>>(xW, e_src, e_dst, e_type,
                                                                  cnt, h1, NEDGE);
  fill_f32<<<dim3(16384), dim3(256), 0, stream>>>(agg, 0.f, (long long)NTOK * HS);
  scatter_agg<<<dim3((NEDGE * 64) / 256), dim3(256), 0, stream>>>(h1, e_src, e_dst, agg, NEDGE);
  castN(h1,  h1bf,  (long long)NTOK * HS);
  castN(agg, aggbf, (long long)NTOK * HS);
  gemm(aggbf, HS, 0, wgrel,  HS, 0, h2, HS, 0, bg_rel, 0, 0, 0, NTOK, HS, HS, 0, 0, 1);
  gemm(h1bf,  HS, 0, wgroot, HS, 0, h2, HS, 0, 0,      0, 0, 0, NTOK, HS, HS, 0, 1, 1);

  // ---- LSTM layer 0 ----
  build_conv<<<dim3((NTOK * DH) / 256), dim3(256), 0, stream>>>(features, h2, convbf);
  gemm(convbf, DH, 0, wih0,  DH, 0, preF, G4, 0, bias0,  0, 0, 0, NTOK, G4, DH, 0, 0, 1);
  gemm(convbf, DH, 0, wih0r, DH, 0, preR, G4, 0, bias0r, 0, 0, 0, NTOK, G4, DH, 0, 0, 1);
  lstm_scan<<<dim3(2), dim3(1024), 0, stream>>>(preF, preR, whh0, whh0r, l0out, TLEN);

  // ---- LSTM layer 1 ----
  castN(l0out, l0bf, (long long)NTOK * MEMD);
  gemm(l0bf, MEMD, 0, wih1,  MEMD, 0, preF, G4, 0, bias1,  0, 0, 0, NTOK, G4, MEMD, 0, 0, 1);
  gemm(l0bf, MEMD, 0, wih1r, MEMD, 0, preR, G4, 0, bias1r, 0, 0, 0, NTOK, G4, MEMD, 0, 0, 1);
  lstm_scan<<<dim3(2), dim3(1024), 0, stream>>>(preF, preR, whh1, whh1r, Mbuf, TLEN);

  // ---- Attention ----
  castN(Mbuf, Mbf, (long long)NTOK * MEMD);
  cast_mask<<<dim3((NTOK * MEMD) / 256), dim3(256), 0, stream>>>(Mbuf, umask, Mmbf);
  build_Ms<<<dim3((BSZ * MEMD * TLEN) / 256), dim3(256), 0, stream>>>(Mbuf, Msbf);
  gemm(Mbf, MEMD, 0, watt, MEMD, 0, Xt, MEMD, 0, b_att, 0, 0, 0,
       NTOK, MEMD, MEMD, 0, 0, 1);                                     // Xt = M@W_att^T + b
  castN(Xt, Xtbf, (long long)NTOK * MEMD);
  // scores[b,t,s] = tanh( (Xt[t,b,:].Mm[s,b,:]) * umask[b,s] ), batched over b
  gemm(Xtbf, BSZ * MEMD, MEMD, Mmbf, BSZ * MEMD, MEMD,
       scores, TLEN, (long long)TLEN * TLEN,
       0, 0, umask, TLEN, TLEN, TLEN, MEMD, 2, 0, BSZ);
  softmax_row<<<dim3(BSZ * TLEN), dim3(512), 0, stream>>>(scores, umask, alphabf);
  // att[t,b,m] = alpha[b,t,:] @ Ms[b,m,:]^T   (C strided back into (T,B,MEM))
  gemm(alphabf, TLEN, (long long)TLEN * TLEN, Msbf, TLEN, (long long)MEMD * TLEN,
       attb, BSZ * MEMD, MEMD, 0, 0, 0, 0, TLEN, MEMD, TLEN, 0, 0, BSZ);
  castN(attb, attbf, (long long)NTOK * MEMD);
  // out = relu(att @ W_lin^T + b_lin)
  gemm(attbf, MEMD, 0, wlin, MEMD, 0, outp, DH, 0, b_lin, 0, 0, 0,
       NTOK, DH, MEMD, 1, 0, 1);
}
